// TransformerEncoder_41085657153697
// MI455X (gfx1250) — matrix operand order repaired and checked by reading, not compiled, not run on hardware
//
#include <hip/hip_runtime.h>
#include <hip/hip_bf16.h>

typedef __attribute__((ext_vector_type(16))) _Float16     v16h;
typedef __attribute__((ext_vector_type(8)))  float        v8f;
typedef __attribute__((ext_vector_type(4)))  int          v4i;
typedef __attribute__((ext_vector_type(4)))  unsigned int v4u;
typedef __attribute__((ext_vector_type(8)))  unsigned int v8u;

#define F_DIM  1024
#define HEADS  16
#define DH     64
#define MLP_H  4096
#define BB     8
#define TT     1024
#define ROWS   (BB * TT)   // 8192

// ---------------------------------------------------------------------------
// CDNA5 async global->LDS path (ASYNCcnt-tracked), guarded by __has_builtin.
// ---------------------------------------------------------------------------
#if defined(__has_builtin)
#  if __has_builtin(__builtin_amdgcn_global_load_async_to_lds_b128)
#    define USE_ASYNC 1
#  else
#    define USE_ASYNC 0
#  endif
#else
#  define USE_ASYNC 0
#endif

typedef __attribute__((address_space(1))) v4i* gv4i_p;
typedef __attribute__((address_space(3))) v4i* lv4i_p;

__device__ __forceinline__ void async_copy_b128(const _Float16* g, _Float16* l) {
#if USE_ASYNC
    __builtin_amdgcn_global_load_async_to_lds_b128((gv4i_p)g, (lv4i_p)l, 0, 0);
#else
    *(float4*)l = *(const float4*)g;
#endif
}

__device__ __forceinline__ void wait_async() {
#if USE_ASYNC
#  if __has_builtin(__builtin_amdgcn_s_wait_asynccnt)
    __builtin_amdgcn_s_wait_asynccnt(0);
#  else
    asm volatile("s_wait_asynccnt 0x0" ::: "memory");
#  endif
#endif
}

__device__ __forceinline__ void wait_tensor() {
#if defined(__has_builtin)
#  if __has_builtin(__builtin_amdgcn_s_wait_tensorcnt)
    __builtin_amdgcn_s_wait_tensorcnt(0);
#  else
    asm volatile("s_wait_tensorcnt 0x0" ::: "memory");
#  endif
#else
    asm volatile("s_wait_tensorcnt 0x0" ::: "memory");
#endif
}

// LDS byte offset of a generic pointer that points into shared memory.
__device__ __forceinline__ unsigned int lds_offset(const void* p) {
    return (unsigned int)(uintptr_t)(__attribute__((address_space(3))) const void*)p;
}

// ---------------------------------------------------------------------------
// Tensor Data Mover: 2D tile load global->LDS (ISA ch.8 D# layout).
// data_size = 2 bytes (f16), tile = 32 (dim0) x 128 (dim1),
// LDS pad: 4 DWORDs (16B) after every 16 DWORDs (64B row) -> stride 40 f16.
// Issued by one wave; tracked by TENSORcnt.
// ---------------------------------------------------------------------------
__device__ __forceinline__ void tdm_load_tile_2d(
    const _Float16* gtile, unsigned int lds_byte, unsigned int tensor_d0,
    unsigned int tensor_d1, unsigned long long stride_elems) {
    const unsigned long long ga = (unsigned long long)(uintptr_t)gtile;
    v4u g0;
    g0[0] = 1u;                                   // count=1, is_restore=0
    g0[1] = lds_byte;                             // lds_addr (bytes)
    g0[2] = (unsigned int)(ga & 0xFFFFFFFFu);     // global_addr[31:0]
    g0[3] = (unsigned int)((ga >> 32) & 0x01FFFFFFu) | (2u << 30);  // [56:32] | type=2
    v8u g1;
    g1[0] = (1u << 16)          // data_size = 2 bytes
          | (1u << 20)          // pad_enable
          | (3u << 22)          // pad_interval: 16 DWORDs
          | (3u << 25);         // pad_amount:   4 DWORDs
    g1[1] = (tensor_d0 & 0xFFFFu) << 16;                       // tensor_dim0[15:0]
    g1[2] = (tensor_d0 >> 16) | ((tensor_d1 & 0xFFFFu) << 16); // dim0 hi | dim1 lo
    g1[3] = (tensor_d1 >> 16) | (32u << 16);                   // dim1 hi | tile_dim0=32
    g1[4] = 128u;                                              // tile_dim1=128, tile_dim2=0
    const unsigned long long sb = stride_elems;                // dim0 stride (elems)
    g1[5] = (unsigned int)(sb & 0xFFFFFFFFu);
    g1[6] = (unsigned int)((sb >> 32) & 0xFFFFu);              // dim1_stride lo16 = 0
    g1[7] = 0u;
    const v4u gz = {0u, 0u, 0u, 0u};
    asm volatile("tensor_load_to_lds %0, %1, %2, %3"
                 :: "s"(g0), "s"(g1), "s"(gz), "s"(gz)
                 : "memory");
}

// ---------------------------------------------------------------------------
// WMMA helpers (gfx1250, wave32). Fragment layouts per CDNA5 ISA §7.12.2.
// ---------------------------------------------------------------------------
__device__ __forceinline__ v8f wmma16(v16h a, v16h b, v8f c) {
    return __builtin_amdgcn_wmma_f32_16x16x32_f16(
        false, a, false, b, (short)0, c, false, false);
}

__device__ __forceinline__ v16h frag_a(const _Float16* base, int ld) {
    const int lane = threadIdx.x & 31;
    const int r    = lane & 15;
    const int hi   = lane >> 4;
    const _Float16* rowp = base + r * ld;
    const int kb0 = hi * 8;
    const int kb1 = 16 + hi * 8;
    v16h a;
#pragma unroll
    for (int j = 0; j < 4; ++j) {
        a[2 * j]     = rowp[kb0 + 2 * j];
        a[2 * j + 1] = rowp[kb0 + 2 * j + 1];
        a[8 + 2 * j]     = rowp[kb1 + 2 * j];
        a[8 + 2 * j + 1] = rowp[kb1 + 2 * j + 1];
    }
    return a;
}

__device__ __forceinline__ v16h frag_b(const _Float16* baseT, int ld) {
    const int lane = threadIdx.x & 31;
    const int n    = lane & 15;
    const int hi   = lane >> 4;
    const _Float16* colp = baseT + n * ld + hi * 8;
    v16h b;
#pragma unroll
    for (int i = 0; i < 16; ++i) b[i] = colp[(i < 8) ? i : (i + 8)];
    return b;
}

__device__ __forceinline__ float gelu_erf(float x) {
    return 0.5f * x * (1.0f + erff(x * 0.70710678118654752440f));
}

// ---------------------------------------------------------------------------
// Weight prep: f32 [K,N] -> f16 [N,K] (transpose via LDS tile), once per call.
// ---------------------------------------------------------------------------
__global__ __launch_bounds__(256) void transpose_cvt_kernel(
    const float* __restrict__ src, _Float16* __restrict__ dst, int K, int N) {
    __shared__ float t[32][33];
    const int bn = blockIdx.x * 32;
    const int bk = blockIdx.y * 32;
    const int tx = threadIdx.x & 31;
    const int ty = threadIdx.x >> 5;
#pragma unroll
    for (int i = 0; i < 4; ++i)
        t[ty + 8 * i][tx] = src[(size_t)(bk + ty + 8 * i) * N + bn + tx];
    __syncthreads();
#pragma unroll
    for (int i = 0; i < 4; ++i)
        dst[(size_t)(bn + ty + 8 * i) * K + bk + tx] = (_Float16)t[tx][ty + 8 * i];
}

// ---------------------------------------------------------------------------
// LayerNorm over last dim (1024) -> f16 output. One block per row.
// ---------------------------------------------------------------------------
__global__ __launch_bounds__(256) void ln_f16_kernel(
    const float* __restrict__ x, const float* __restrict__ g,
    const float* __restrict__ b, _Float16* __restrict__ out) {
    __shared__ float red[256];
    const int row = blockIdx.x;
    const float* xr = x + (size_t)row * F_DIM;
    float v[4];
    float s = 0.f;
#pragma unroll
    for (int i = 0; i < 4; ++i) { v[i] = xr[threadIdx.x + 256 * i]; s += v[i]; }
    red[threadIdx.x] = s;
    __syncthreads();
    for (int o = 128; o > 0; o >>= 1) {
        if (threadIdx.x < o) red[threadIdx.x] += red[threadIdx.x + o];
        __syncthreads();
    }
    const float mu = red[0] * (1.0f / F_DIM);
    __syncthreads();
    float s2 = 0.f;
#pragma unroll
    for (int i = 0; i < 4; ++i) { float d = v[i] - mu; s2 += d * d; }
    red[threadIdx.x] = s2;
    __syncthreads();
    for (int o = 128; o > 0; o >>= 1) {
        if (threadIdx.x < o) red[threadIdx.x] += red[threadIdx.x + o];
        __syncthreads();
    }
    const float rstd = rsqrtf(red[0] * (1.0f / F_DIM) + 1e-5f);
    _Float16* orow = out + (size_t)row * F_DIM;
#pragma unroll
    for (int i = 0; i < 4; ++i) {
        const int c = threadIdx.x + 256 * i;
        orow[c] = (_Float16)((v[i] - mu) * rstd * g[c] + b[c]);
    }
}

// ---------------------------------------------------------------------------
// WMMA GEMM: C[M,N] = A[M,K](f16,row-major) * Bt[N,K](f16,N-major) + epilogue
// 128x128 block tile, BK=32, 8 waves, wave tile 64x32 (8 WMMA / wave / step).
// Double-buffered LDS; tiles staged by the Tensor Data Mover (wave 0 posts
// two 2D D# descriptors per K-step; LDS row padding done by TDM pad feature).
// ---------------------------------------------------------------------------
enum { EPI_BIAS_F16 = 0, EPI_BIAS_RES_F32 = 1, EPI_BIAS_GELU_F16 = 2,
       EPI_BIAS_GELU_RES_F32 = 3 };

template <int EPI>
__global__ __launch_bounds__(256) void gemm_wmma_kernel(
    const _Float16* __restrict__ A, const _Float16* __restrict__ Bt,
    const float* __restrict__ bias, const float* __restrict__ resid,
    _Float16* __restrict__ outH, float* __restrict__ outF,
    int M, int N, int K) {
    __shared__ _Float16 sA[2][128][40];
    __shared__ _Float16 sB[2][128][40];

    const int tid  = threadIdx.x;
    const int lane = tid & 31;
    const int wave = tid >> 5;
    const int wr = wave >> 2;
    const int wc = wave & 3;
    const int bm = blockIdx.x * 128;
    const int bn = blockIdx.y * 128;

    v8f acc[4][2];
#pragma unroll
    for (int i = 0; i < 4; ++i)
#pragma unroll
        for (int j = 0; j < 2; ++j) acc[i][j] = (v8f){};

    // prologue: TDM-stage K-tile 0 into buffer 0 (wave 0 posts descriptors)
    if (wave == 0) {
        tdm_load_tile_2d(A  + (size_t)bm * K, lds_offset(&sA[0][0][0]),
                         (unsigned)K, (unsigned)M, (unsigned long long)K);
        tdm_load_tile_2d(Bt + (size_t)bn * K, lds_offset(&sB[0][0][0]),
                         (unsigned)K, (unsigned)N, (unsigned long long)K);
        wait_tensor();
    }
    __syncthreads();

    int buf = 0;
    for (int k0 = 0; k0 < K; k0 += 32) {
        const int nxt = k0 + 32;
        if (nxt < K && wave == 0) {  // stream next tile into the other buffer
            const int nb = buf ^ 1;
            tdm_load_tile_2d(A  + (size_t)bm * K + nxt, lds_offset(&sA[nb][0][0]),
                             (unsigned)K, (unsigned)M, (unsigned long long)K);
            tdm_load_tile_2d(Bt + (size_t)bn * K + nxt, lds_offset(&sB[nb][0][0]),
                             (unsigned)K, (unsigned)N, (unsigned long long)K);
        }
        const v16h b0 = frag_b(&sB[buf][wc * 32][0], 40);
        const v16h b1 = frag_b(&sB[buf][wc * 32 + 16][0], 40);
#pragma unroll
        for (int i = 0; i < 4; ++i) {
            const v16h a = frag_a(&sA[buf][wr * 64 + i * 16][0], 40);
            acc[i][0] = wmma16(a, b0, acc[i][0]);
            acc[i][1] = wmma16(a, b1, acc[i][1]);
        }
        if (wave == 0) wait_tensor();
        __syncthreads();
        buf ^= 1;
    }

    // epilogue
    const int cl = lane & 15;
    const int rTop = bm + wr * 64 + (lane >> 4) * 8;
#pragma unroll
    for (int j = 0; j < 2; ++j) {
        const int col = bn + wc * 32 + j * 16 + cl;
        const float bb = bias[col];
#pragma unroll
        for (int i = 0; i < 4; ++i) {
#pragma unroll
            for (int v = 0; v < 8; ++v) {
                const int row = rTop + i * 16 + v;
                float xv = acc[i][j][v] + bb;
                if constexpr (EPI == EPI_BIAS_GELU_F16 || EPI == EPI_BIAS_GELU_RES_F32)
                    xv = gelu_erf(xv);
                if constexpr (EPI == EPI_BIAS_RES_F32 || EPI == EPI_BIAS_GELU_RES_F32) {
                    const size_t idx = (size_t)row * N + col;
                    outF[idx] = xv + resid[idx];
                } else {
                    outH[(size_t)row * N + col] = (_Float16)xv;
                }
            }
        }
    }
}

// ---------------------------------------------------------------------------
// Flash attention: grid (B*HEADS, T/64). Q/K/V stored [B*T, F] f16, head is a
// strided slice (stride F_DIM). Online softmax, scale = 1/sqrt(F) = 1/32.
// Uses async global->LDS staging (ASYNCcnt path).
// ---------------------------------------------------------------------------
__global__ __launch_bounds__(256) void attn_kernel(
    const _Float16* __restrict__ Q, const _Float16* __restrict__ Kk,
    const _Float16* __restrict__ V, _Float16* __restrict__ O) {
    __shared__ _Float16 sQ[64][72];
    __shared__ _Float16 sK[64][72];
    __shared__ _Float16 sVt[64][72];
    __shared__ float    sS[64][65];
    __shared__ _Float16 sP[64][72];
    __shared__ float mRow[64], lRow[64], cRow[64];

    const int tid  = threadIdx.x;
    const int lane = tid & 31;
    const int wave = tid >> 5;
    const int wm = wave & 3;
    const int wn = wave >> 2;
    const int bh = blockIdx.x;
    const int b  = bh / HEADS;
    const int h  = bh % HEADS;
    const int q0 = blockIdx.y * 64;
    const size_t base = (size_t)b * TT * F_DIM + (size_t)h * DH;

    const int lr = tid >> 2;
    const int lc = (tid & 3) * 16;

    {   // stage Q tile (async)
        const _Float16* qp = Q + base + (size_t)(q0 + lr) * F_DIM + lc;
        async_copy_b128(qp,     &sQ[lr][lc]);
        async_copy_b128(qp + 8, &sQ[lr][lc + 8]);
    }
    if (tid < 64) { mRow[tid] = -1e30f; lRow[tid] = 0.f; }
    v8f o0 = {}, o1 = {};
    wait_async();
    __syncthreads();

    for (int j = 0; j < TT / 64; ++j) {
        {   // stage K tile (async) + V tile transposed (manual)
            const size_t krow = base + (size_t)(j * 64 + lr) * F_DIM + lc;
            const _Float16* kp = Kk + krow;
            async_copy_b128(kp,     &sK[lr][lc]);
            async_copy_b128(kp + 8, &sK[lr][lc + 8]);
            const float4* vs = (const float4*)(V + krow);
            float4 v0 = vs[0], v1 = vs[1];
            const _Float16* vh0 = (const _Float16*)&v0;
            const _Float16* vh1 = (const _Float16*)&v1;
#pragma unroll
            for (int i = 0; i < 8; ++i) {
                sVt[lc + i][lr]     = vh0[i];
                sVt[lc + 8 + i][lr] = vh1[i];
            }
        }
        wait_async();
        __syncthreads();

        // S = Q * K^T
        v8f s0 = {}, s1 = {};
#pragma unroll
        for (int kk = 0; kk < 64; kk += 32) {
            const v16h a   = frag_a(&sQ[wm * 16][kk], 72);
            const v16h bb0 = frag_b(&sK[wn * 32][kk], 72);
            const v16h bb1 = frag_b(&sK[wn * 32 + 16][kk], 72);
            s0 = wmma16(a, bb0, s0);
            s1 = wmma16(a, bb1, s1);
        }
        {   // spill scaled scores
            const float sc = 0.03125f;  // 1/sqrt(1024)
            const int cc = wn * 32 + (lane & 15);
            const int rb = wm * 16 + (lane >> 4) * 8;
#pragma unroll
            for (int v = 0; v < 8; ++v) {
                sS[rb + v][cc]      = s0[v] * sc;
                sS[rb + v][cc + 16] = s1[v] * sc;
            }
        }
        __syncthreads();

        if (tid < 64) {  // online softmax row update
            float mx = mRow[tid];
#pragma unroll
            for (int c = 0; c < 64; ++c) mx = fmaxf(mx, sS[tid][c]);
            const float scl = __expf(mRow[tid] - mx);
            float sum = 0.f;
#pragma unroll
            for (int c = 0; c < 64; ++c) {
                const float p = __expf(sS[tid][c] - mx);
                sP[tid][c] = (_Float16)p;
                sum += p;
            }
            mRow[tid] = mx;
            lRow[tid] = lRow[tid] * scl + sum;
            cRow[tid] = scl;
        }
        __syncthreads();

        {   // rescale running O
            const int rb = wm * 16 + (lane >> 4) * 8;
#pragma unroll
            for (int v = 0; v < 8; ++v) {
                const float s = cRow[rb + v];
                o0[v] *= s;
                o1[v] *= s;
            }
        }
        // O += P * V
#pragma unroll
        for (int kk = 0; kk < 64; kk += 32) {
            const v16h a   = frag_a(&sP[wm * 16][kk], 72);
            const v16h bb0 = frag_b(&sVt[wn * 32][kk], 72);
            const v16h bb1 = frag_b(&sVt[wn * 32 + 16][kk], 72);
            o0 = wmma16(a, bb0, o0);
            o1 = wmma16(a, bb1, o1);
        }
        __syncthreads();
    }

    {   // normalize and store
        const int cc = wn * 32 + (lane & 15);
        const int rb = wm * 16 + (lane >> 4) * 8;
#pragma unroll
        for (int v = 0; v < 8; ++v) {
            const int row = rb + v;
            const float inv = 1.0f / lRow[row];
            O[base + (size_t)(q0 + row) * F_DIM + cc]      = (_Float16)(o0[v] * inv);
            O[base + (size_t)(q0 + row) * F_DIM + cc + 16] = (_Float16)(o1[v] * inv);
        }
    }
}

// ---------------------------------------------------------------------------
// Host-side orchestration
// ---------------------------------------------------------------------------
extern "C" void kernel_launch(void* const* d_in, const int* in_sizes, int n_in,
                              void* d_out, int out_size, void* d_ws, size_t ws_size,
                              hipStream_t stream) {
    (void)in_sizes; (void)n_in; (void)out_size; (void)ws_size;
    const float* x    = (const float*)d_in[0];
    const float* wq   = (const float*)d_in[1];
    const float* bq   = (const float*)d_in[2];
    const float* wk   = (const float*)d_in[3];
    const float* bk   = (const float*)d_in[4];
    const float* wv   = (const float*)d_in[5];
    const float* bv   = (const float*)d_in[6];
    const float* wo   = (const float*)d_in[7];
    const float* bo   = (const float*)d_in[8];
    const float* ln1g = (const float*)d_in[9];
    const float* ln1b = (const float*)d_in[10];
    const float* ln2g = (const float*)d_in[11];
    const float* ln2b = (const float*)d_in[12];
    const float* w1   = (const float*)d_in[13];
    const float* b1   = (const float*)d_in[14];
    const float* w2   = (const float*)d_in[15];
    const float* b2   = (const float*)d_in[16];

    char* ws = (char*)d_ws;
    size_t off = 0;
    auto alloc = [&](size_t bytes) -> void* {
        void* p = ws + off;
        off = (off + bytes + 255) & ~(size_t)255;
        return p;
    };

    _Float16* wqT = (_Float16*)alloc((size_t)F_DIM * F_DIM * 2);
    _Float16* wkT = (_Float16*)alloc((size_t)F_DIM * F_DIM * 2);
    _Float16* wvT = (_Float16*)alloc((size_t)F_DIM * F_DIM * 2);
    _Float16* woT = (_Float16*)alloc((size_t)F_DIM * F_DIM * 2);
    _Float16* w1T = (_Float16*)alloc((size_t)F_DIM * MLP_H * 2);
    _Float16* w2T = (_Float16*)alloc((size_t)MLP_H * F_DIM * 2);
    _Float16* hH    = (_Float16*)alloc((size_t)ROWS * F_DIM * 2);
    _Float16* qH    = (_Float16*)alloc((size_t)ROWS * F_DIM * 2);
    _Float16* kH    = (_Float16*)alloc((size_t)ROWS * F_DIM * 2);
    _Float16* vH    = (_Float16*)alloc((size_t)ROWS * F_DIM * 2);
    _Float16* attnH = (_Float16*)alloc((size_t)ROWS * F_DIM * 2);
    float*    res1  = (float*)alloc((size_t)ROWS * F_DIM * 4);
    _Float16* h2H   = (_Float16*)alloc((size_t)ROWS * F_DIM * 2);
    _Float16* m1H   = (_Float16*)alloc((size_t)ROWS * MLP_H * 2);

    dim3 gT(F_DIM / 32, F_DIM / 32);
    transpose_cvt_kernel<<<gT, 256, 0, stream>>>(wq, wqT, F_DIM, F_DIM);
    transpose_cvt_kernel<<<gT, 256, 0, stream>>>(wk, wkT, F_DIM, F_DIM);
    transpose_cvt_kernel<<<gT, 256, 0, stream>>>(wv, wvT, F_DIM, F_DIM);
    transpose_cvt_kernel<<<gT, 256, 0, stream>>>(wo, woT, F_DIM, F_DIM);
    transpose_cvt_kernel<<<dim3(MLP_H / 32, F_DIM / 32), 256, 0, stream>>>(w1, w1T, F_DIM, MLP_H);
    transpose_cvt_kernel<<<dim3(F_DIM / 32, MLP_H / 32), 256, 0, stream>>>(w2, w2T, MLP_H, F_DIM);

    ln_f16_kernel<<<ROWS, 256, 0, stream>>>(x, ln1g, ln1b, hH);

    dim3 gQ(ROWS / 128, F_DIM / 128);
    gemm_wmma_kernel<EPI_BIAS_F16><<<gQ, 256, 0, stream>>>(
        hH, wqT, bq, nullptr, qH, nullptr, ROWS, F_DIM, F_DIM);
    gemm_wmma_kernel<EPI_BIAS_F16><<<gQ, 256, 0, stream>>>(
        hH, wkT, bk, nullptr, kH, nullptr, ROWS, F_DIM, F_DIM);
    gemm_wmma_kernel<EPI_BIAS_F16><<<gQ, 256, 0, stream>>>(
        hH, wvT, bv, nullptr, vH, nullptr, ROWS, F_DIM, F_DIM);

    attn_kernel<<<dim3(BB * HEADS, TT / 64), 256, 0, stream>>>(qH, kH, vH, attnH);

    gemm_wmma_kernel<EPI_BIAS_RES_F32><<<gQ, 256, 0, stream>>>(
        attnH, woT, bo, x, nullptr, res1, ROWS, F_DIM, F_DIM);

    ln_f16_kernel<<<ROWS, 256, 0, stream>>>(res1, ln2g, ln2b, h2H);

    dim3 gM1(ROWS / 128, MLP_H / 128);
    gemm_wmma_kernel<EPI_BIAS_GELU_F16><<<gM1, 256, 0, stream>>>(
        h2H, w1T, b1, nullptr, m1H, nullptr, ROWS, MLP_H, F_DIM);
    dim3 gM2(ROWS / 128, F_DIM / 128);
    gemm_wmma_kernel<EPI_BIAS_GELU_RES_F32><<<gM2, 256, 0, stream>>>(
        m1H, w2T, b2, res1, nullptr, (float*)d_out, ROWS, F_DIM, MLP_H);
}
